// NTXentLoss_9483287789816
// MI455X (gfx1250) — matrix operand order repaired and checked by reading, not compiled, not run on hardware
//
#include <hip/hip_runtime.h>

typedef __attribute__((ext_vector_type(16))) _Float16 v16h;
typedef __attribute__((ext_vector_type(8)))  _Float16 v8h;
typedef __attribute__((ext_vector_type(8)))  float    v8f;
typedef __attribute__((ext_vector_type(4)))  float    v4f;
typedef __attribute__((ext_vector_type(4)))  int      v4i;

#define N_HALF   8192
#define TWO_N    16384
#define D        256
#define INV_T    10.0f
#define EPS_F    1e-6f

#define TILE_PITCH_H 264              // 256 halves + 8 pad halves (528 B row pitch)
#define TILE_COLS    32               // two 16-col WMMA tiles per iteration
#define NTI          (TWO_N / TILE_COLS)   // 512 iterations

#if defined(__gfx1250__) && __has_builtin(__builtin_amdgcn_global_load_async_to_lds_b128)
#define HAVE_ASYNC 1
#else
#define HAVE_ASYNC 0
#endif

#if defined(__gfx1250__) && __has_builtin(__builtin_amdgcn_sched_barrier)
#define SCHED_FENCE() __builtin_amdgcn_sched_barrier(0)
#else
#define SCHED_FENCE()
#endif

// ---------------------------------------------------------------------------
// Kernel 1: row-normalize z = concat(z1,z2) into f16 workspace.
// One wave per row; also zeroes the scalar output for atomic accumulation.
// ---------------------------------------------------------------------------
__global__ void __launch_bounds__(256) ntxent_normalize(
    const float* __restrict__ z1, const float* __restrict__ z2,
    _Float16* __restrict__ zn, float* __restrict__ out)
{
    if (blockIdx.x == 0 && threadIdx.x == 0) out[0] = 0.0f;

    const int row  = blockIdx.x * 8 + (threadIdx.x >> 5);
    const int lane = threadIdx.x & 31;
    if (row >= TWO_N) return;

    const float* src = (row < N_HALF) ? (z1 + (size_t)row * D)
                                      : (z2 + (size_t)(row - N_HALF) * D);
    v4f a = *(const v4f*)(src + lane * 8);
    v4f b = *(const v4f*)(src + lane * 8 + 4);

    float ss = a.x*a.x + a.y*a.y + a.z*a.z + a.w*a.w
             + b.x*b.x + b.y*b.y + b.z*b.z + b.w*b.w;
    #pragma unroll
    for (int m = 16; m >= 1; m >>= 1) ss += __shfl_xor(ss, m, 32);

    const float inv = 1.0f / fmaxf(sqrtf(ss), EPS_F);

    v8h h;
    h[0] = (_Float16)(a.x * inv); h[1] = (_Float16)(a.y * inv);
    h[2] = (_Float16)(a.z * inv); h[3] = (_Float16)(a.w * inv);
    h[4] = (_Float16)(b.x * inv); h[5] = (_Float16)(b.y * inv);
    h[6] = (_Float16)(b.z * inv); h[7] = (_Float16)(b.w * inv);
    *(v8h*)(zn + (size_t)row * D + lane * 8) = h;
}

// ---------------------------------------------------------------------------
// Async DMA of one 32-col B tile (32 rows x 512 B) into padded LDS.
// 128 threads x 8 b128 transfers; INST_OFFSET applies to both addresses,
// and all 8 chunks stay inside one row, so padding is preserved.
// ---------------------------------------------------------------------------
#if HAVE_ASYNC
__device__ __forceinline__ void issue_tile_load(
    const _Float16* __restrict__ zn, int colBase, _Float16* ldsBuf, int tid)
{
    const int row = tid >> 2;      // 0..31
    const int seg = tid & 3;       // 0..3 (128 B chunks)
    const _Float16* g = zn + (size_t)(colBase + row) * D + seg * 64;
    _Float16*       l = ldsBuf + row * TILE_PITCH_H + seg * 64;
    auto gp = (__attribute__((address_space(1))) v4i*)g;   // C-cast drops const
    auto lp = (__attribute__((address_space(3))) v4i*)l;
    __builtin_amdgcn_global_load_async_to_lds_b128(gp, lp, 0,   0);
    __builtin_amdgcn_global_load_async_to_lds_b128(gp, lp, 16,  0);
    __builtin_amdgcn_global_load_async_to_lds_b128(gp, lp, 32,  0);
    __builtin_amdgcn_global_load_async_to_lds_b128(gp, lp, 48,  0);
    __builtin_amdgcn_global_load_async_to_lds_b128(gp, lp, 64,  0);
    __builtin_amdgcn_global_load_async_to_lds_b128(gp, lp, 80,  0);
    __builtin_amdgcn_global_load_async_to_lds_b128(gp, lp, 96,  0);
    __builtin_amdgcn_global_load_async_to_lds_b128(gp, lp, 112, 0);
}

__device__ __forceinline__ void wait_async0()
{
#if __has_builtin(__builtin_amdgcn_s_wait_asynccnt)
    __builtin_amdgcn_s_wait_asynccnt(0);
#else
    asm volatile("s_wait_asynccnt 0x0" ::: "memory");
#endif
}
#endif

// ---------------------------------------------------------------------------
// Kernel 2: fused sim = Zn @ Zn^T with per-row masked logsumexp + positives.
// Wave = 16-row strip (A panel resident in VGPRs) x 32 columns per step via
// TWO independent WMMA chains. sched_barrier(0) fences force batched
// ds_load groups ahead of WMMA bursts so LDS latency is paid once (pipelined)
// instead of once per WMMA; the K4-7 loads are scheduled UNDER the K0-3
// WMMA burst. Block of 4 waves shares a double-buffered LDS B tile filled by
// async global->LDS DMA. Logits bounded by 1/T = 10 -> fixed-max LSE.
// ---------------------------------------------------------------------------
__global__ void __launch_bounds__(128) ntxent_gemm_lse(
    const _Float16* __restrict__ zn, float* __restrict__ out)
{
    const int tid     = threadIdx.x;
    const int lane    = tid & 31;
    const int wave    = tid >> 5;                 // 0..3
    const int rowBase = blockIdx.x * 64 + wave * 16;
    const int half    = lane >> 4;                // 0 or 1
    const int l15     = lane & 15;

    // ---- hoist A panel (contiguous k = 16*half + s per 32-k step, as B) ----------------
    const _Float16* aRow = zn + (size_t)(rowBase + l15) * D + half * 16;
    v16h a[8];
    #pragma unroll
    for (int k8 = 0; k8 < 8; ++k8) {
        v8h lo = *(const v8h*)(aRow + k8 * 32);
        v8h hi = *(const v8h*)(aRow + k8 * 32 + 8);
        a[k8] = __builtin_shufflevector(lo, hi, 0,1,2,3,4,5,6,7,
                                                8,9,10,11,12,13,14,15);
    }

    float s[8], pos[8];
    int rowId[8], posCol[8];
    #pragma unroll
    for (int r = 0; r < 8; ++r) {
        s[r] = 0.0f; pos[r] = 0.0f;
        rowId[r]  = rowBase + r + half * 8;
        posCol[r] = (rowId[r] < N_HALF) ? rowId[r] + N_HALF : rowId[r] - N_HALF;
    }

#if HAVE_ASYNC
    __shared__ __align__(16) _Float16 ldsT[2][TILE_COLS * TILE_PITCH_H];
    issue_tile_load(zn, 0, ldsT[0], tid);
    wait_async0();
    __syncthreads();
#endif

    for (int ct = 0; ct < NTI; ++ct) {
        const int col0 = ct * TILE_COLS + l15;
        const int col1 = col0 + 16;

        v8f c0 = {}, c1 = {};
#if HAVE_ASYNC
        const int cur = ct & 1;
        const int nxt = (ct + 1 < NTI) ? ct + 1 : ct;   // branchless tail
        issue_tile_load(zn, nxt * TILE_COLS, ldsT[cur ^ 1], tid);

        // conflict-free LDS reads: banks (4*l15 + 8*half + 16*k8) mod 64;
        // +16 rows = 8448 B == 0 (mod 256 B) -> same conflict-free pattern
        const _Float16* bp0 = &ldsT[cur][l15 * TILE_PITCH_H + half * 16];
        const _Float16* bp1 = bp0 + 16 * TILE_PITCH_H;
#else
        const _Float16* bp0 = zn + (size_t)col0 * D + half * 16;
        const _Float16* bp1 = zn + (size_t)col1 * D + half * 16;
#endif
        v16h p0[4], p1[4], q0[4], q1[4];
        #pragma unroll
        for (int j = 0; j < 4; ++j) {               // K-steps 0..3 loads
            p0[j] = *(const v16h*)(bp0 + j * 32);
            p1[j] = *(const v16h*)(bp1 + j * 32);
        }
        SCHED_FENCE();
        #pragma unroll
        for (int j = 0; j < 4; ++j) {               // K-steps 4..7 loads:
            q0[j] = *(const v16h*)(bp0 + (j + 4) * 32); // scheduled under
            q1[j] = *(const v16h*)(bp1 + (j + 4) * 32); // the WMMAs below
        }
        #pragma unroll
        for (int j = 0; j < 4; ++j) {               // K-steps 0..3 WMMAs
            c0 = __builtin_amdgcn_wmma_f32_16x16x32_f16(
                     false, a[j], false, p0[j], (short)0, c0, false, false);
            c1 = __builtin_amdgcn_wmma_f32_16x16x32_f16(
                     false, a[j], false, p1[j], (short)0, c1, false, false);
        }
        SCHED_FENCE();
        #pragma unroll
        for (int j = 0; j < 4; ++j) {               // K-steps 4..7 WMMAs
            c0 = __builtin_amdgcn_wmma_f32_16x16x32_f16(
                     false, a[j + 4], false, q0[j], (short)0, c0, false, false);
            c1 = __builtin_amdgcn_wmma_f32_16x16x32_f16(
                     false, a[j + 4], false, q1[j], (short)0, c1, false, false);
        }

        #pragma unroll
        for (int r = 0; r < 8; ++r) {
            float v0 = fminf(fmaxf(c0[r], -1.0f + EPS_F), 1.0f - EPS_F);
            float v1 = fminf(fmaxf(c1[r], -1.0f + EPS_F), 1.0f - EPS_F);
            float e0 = __expf(v0 * INV_T - INV_T);       // exp(logit - 10)
            float e1 = __expf(v1 * INV_T - INV_T);
            s[r]   += (col0 == rowId[r])  ? 0.0f : e0;   // mask diagonal
            s[r]   += (col1 == rowId[r])  ? 0.0f : e1;
            pos[r] += (col0 == posCol[r]) ? v0   : 0.0f; // capture positive
            pos[r] += (col1 == posCol[r]) ? v1   : 0.0f;
        }

#if HAVE_ASYNC
        wait_async0();      // next tile landed (hidden behind 16 WMMAs+epilogue)
        __syncthreads();    // all waves done reading cur, DMA visible
#endif
    }

    // ---- reduce across the 16 lanes of each half (wave32 butterflies) ----
    #pragma unroll
    for (int r = 0; r < 8; ++r) {
        #pragma unroll
        for (int m = 1; m <= 8; m <<= 1) {
            s[r]   += __shfl_xor(s[r],   m, 32);
            pos[r] += __shfl_xor(pos[r], m, 32);
        }
    }

    float acc = 0.0f;
    #pragma unroll
    for (int r = 0; r < 8; ++r)
        acc += -pos[r] * INV_T + (INV_T + __logf(s[r]));

    acc += __shfl_xor(acc, 16, 32);      // combine halves
    if (lane == 0)
        atomicAdd(out, acc * (1.0f / (float)TWO_N));
}

// ---------------------------------------------------------------------------
extern "C" void kernel_launch(void* const* d_in, const int* in_sizes, int n_in,
                              void* d_out, int out_size, void* d_ws, size_t ws_size,
                              hipStream_t stream)
{
    (void)in_sizes; (void)n_in; (void)out_size; (void)ws_size;
    const float* z1 = (const float*)d_in[0];
    const float* z2 = (const float*)d_in[1];
    float* out      = (float*)d_out;
    _Float16* zn    = (_Float16*)d_ws;      // 16384*256*2B = 8 MB scratch

    ntxent_normalize<<<TWO_N / 8, 256, 0, stream>>>(z1, z2, zn, out);
    ntxent_gemm_lse<<<TWO_N / 64, 128, 0, stream>>>(zn, out);
}
